// FeedForwardQuantum_65481071407547
// MI455X (gfx1250) — matrix operand order repaired and checked by reading, not compiled, not run on hardware
//
#include <hip/hip_runtime.h>

typedef __attribute__((ext_vector_type(16))) _Float16 v16h;
typedef __attribute__((ext_vector_type(8)))  _Float16 v8h;
typedef __attribute__((ext_vector_type(4)))  _Float16 v4h;
typedef __attribute__((ext_vector_type(8)))  float    v8f;

#define BQ      8
#define SQ      2048
#define EMBED   1024
#define FFN     4096
#define NQ      8
#define M_TOTAL (BQ * SQ)          /* 16384 tokens */
#define M_TILE  32                 /* tokens per block */
#define LDH     (FFN + 8)          /* padded LDS row stride in halfs (16B pad) */

// ---------------------------------------------------------------------------
// Kernel 1: W2 f32 [EMBED][FFN] -> f16 [EMBED][FFN] in workspace.
// Row-major f16 W2 gives B^T in this kernel's contiguous k order: lane (n, halfk)
// reads 32 contiguous bytes of row n at K offset 16*halfk.
// ---------------------------------------------------------------------------
__global__ __launch_bounds__(256) void prep_w2_f16(const float* __restrict__ W2,
                                                   _Float16* __restrict__ w2h) {
  int idx = (blockIdx.x * 256 + threadIdx.x) * 4;
  float4 v = *reinterpret_cast<const float4*>(W2 + idx);
  v4h o;
  o[0] = (_Float16)v.x;
  o[1] = (_Float16)v.y;
  o[2] = (_Float16)v.z;
  o[3] = (_Float16)v.w;
  *reinterpret_cast<v4h*>(w2h + idx) = o;
}

// ---------------------------------------------------------------------------
// Kernel 2: fused  q -> h (LDS, f16) -> out = h @ W2^T + b2  via WMMA.
// ---------------------------------------------------------------------------
__global__ __launch_bounds__(256) void ffq_fused(
    const float* __restrict__ x,     const float* __restrict__ theta,
    const float* __restrict__ W1,    const float* __restrict__ b1,
    const _Float16* __restrict__ w2h, const float* __restrict__ b2,
    float* __restrict__ out) {
  extern __shared__ char smem[];
  _Float16* hbuf = reinterpret_cast<_Float16*>(smem);                       // [M_TILE][LDH] f16
  float*    qbuf = reinterpret_cast<float*>(smem + (size_t)M_TILE * LDH * sizeof(_Float16)); // [M_TILE][NQ]

  const int tid = threadIdx.x;
  const int m0  = blockIdx.x * M_TILE;

  // ---- phase 0: q = cos(x[:, :NQ]) * cos(theta) ------------------------------
  {
    const int m = tid >> 3;          // 0..31
    const int i = tid & 7;           // 0..7
    const float xv = x[(size_t)(m0 + m) * EMBED + i];
    qbuf[tid] = __cosf(xv) * __cosf(theta[i]);
  }
  __syncthreads();

  // ---- phase 1: h = relu(q @ W1^T + b1) -> f16 in LDS ------------------------
  // Thread keeps its W1 row (8 floats) in registers, sweeps 32 tokens.
  for (int j = 0; j < FFN / 256; ++j) {
    const int f = tid + j * 256;
    const float4 wa = *reinterpret_cast<const float4*>(W1 + f * NQ);
    const float4 wb = *reinterpret_cast<const float4*>(W1 + f * NQ + 4);
    const float bias = b1[f];
#pragma unroll 4
    for (int m = 0; m < M_TILE; ++m) {
      const float* q = qbuf + m * NQ;
      float s = bias;
      s = fmaf(q[0], wa.x, s); s = fmaf(q[1], wa.y, s);
      s = fmaf(q[2], wa.z, s); s = fmaf(q[3], wa.w, s);
      s = fmaf(q[4], wb.x, s); s = fmaf(q[5], wb.y, s);
      s = fmaf(q[6], wb.z, s); s = fmaf(q[7], wb.w, s);
      s = fmaxf(s, 0.0f);
      hbuf[m * LDH + f] = (_Float16)s;
    }
  }
  __syncthreads();

  // ---- phase 2: out[m0:m0+32, :] = h @ W2^T + b2 -----------------------------
  const int lane  = tid & 31;
  const int wave  = tid >> 5;        // 0..7, each wave owns a 128-col slice
  const int nrow  = lane & 15;       // row within A-tile / col within B-tile
  const int halfk = lane >> 4;       // K-half selector per ISA layouts
  const int n_base = wave * 128;

  v8f acc[2][8];
#pragma unroll
  for (int mt = 0; mt < 2; ++mt)
#pragma unroll
    for (int nt = 0; nt < 8; ++nt)
#pragma unroll
      for (int e = 0; e < 8; ++e) acc[mt][nt][e] = 0.0f;

  const _Float16* arow[2] = { hbuf + nrow * LDH, hbuf + (16 + nrow) * LDH };
  // Per-lane base of this wave's B-fragment bytes (col n = nrow of each n-tile).
  const _Float16* bbase = w2h + (size_t)(n_base + nrow) * FFN + 16 * halfk;

  for (int k0 = 0; k0 < FFN; k0 += 32) {
    // ---- B fragments: batch ALL eight 32B loads into distinct registers so
    // only one L2 round-trip is exposed per k-step (waits become descending
    // s_wait_loadcnt overlapping with WMMA issue), and prefetch next k-step.
    v16h bf[8];
#pragma unroll
    for (int nt = 0; nt < 8; ++nt) {
      const _Float16* bp = bbase + (size_t)nt * 16 * FFN + k0;
      bf[nt] = *reinterpret_cast<const v16h*>(bp);
      __builtin_prefetch(bp + 32, 0, 3);   // warm next iteration (global_prefetch)
    }

    // ---- A fragments (16x32 f16): lanes 0-15 hold K {0..15}, lanes
    // 16-31 hold K {16..31} of row (lane&15) -> two 16B LDS chunks each.
    v16h a[2];
#pragma unroll
    for (int mt = 0; mt < 2; ++mt) {
      const v8h c0 = *reinterpret_cast<const v8h*>(arow[mt] + k0 + 16 * halfk);
      const v8h c1 = *reinterpret_cast<const v8h*>(arow[mt] + k0 + 8 + 16 * halfk);
#pragma unroll
      for (int e = 0; e < 8; ++e) { a[mt][e] = c0[e]; a[mt][8 + e] = c1[e]; }
    }

    // ---- 16 WMMAs per k-step; each B fragment feeds both M-tiles.
#pragma unroll
    for (int nt = 0; nt < 8; ++nt) {
      acc[0][nt] = __builtin_amdgcn_wmma_f32_16x16x32_f16(
          false, a[0], false, bf[nt], (short)0, acc[0][nt], false, false);
      acc[1][nt] = __builtin_amdgcn_wmma_f32_16x16x32_f16(
          false, a[1], false, bf[nt], (short)0, acc[1][nt], false, false);
    }
  }

  // ---- epilogue: + b2, store (C layout: VGPR r -> row 8*halfk + r) -----------
#pragma unroll
  for (int nt = 0; nt < 8; ++nt) {
    const int col = n_base + nt * 16 + nrow;
    const float bias = b2[col];
#pragma unroll
    for (int mt = 0; mt < 2; ++mt) {
#pragma unroll
      for (int r = 0; r < 8; ++r) {
        const int row = m0 + mt * 16 + 8 * halfk + r;
        out[(size_t)row * EMBED + col] = acc[mt][nt][r] + bias;
      }
    }
  }
}

// ---------------------------------------------------------------------------
extern "C" void kernel_launch(void* const* d_in, const int* in_sizes, int n_in,
                              void* d_out, int out_size, void* d_ws, size_t ws_size,
                              hipStream_t stream) {
  const float* x     = (const float*)d_in[0];
  const float* theta = (const float*)d_in[1];
  const float* W1    = (const float*)d_in[2];
  const float* b1    = (const float*)d_in[3];
  const float* W2    = (const float*)d_in[4];
  const float* b2    = (const float*)d_in[5];
  float*       out   = (float*)d_out;
  _Float16*    w2h   = (_Float16*)d_ws;    // 8 MB f16 copy of W2

  // W2 -> f16 (4 elems/thread)
  prep_w2_f16<<<(EMBED * FFN) / (256 * 4), 256, 0, stream>>>(W2, w2h);

  const size_t lds_bytes =
      (size_t)M_TILE * LDH * sizeof(_Float16) + (size_t)M_TILE * NQ * sizeof(float);
  (void)hipFuncSetAttribute(reinterpret_cast<const void*>(ffq_fused),
                            hipFuncAttributeMaxDynamicSharedMemorySize,
                            (int)lds_bytes);
  ffq_fused<<<M_TOTAL / M_TILE, 256, lds_bytes, stream>>>(
      x, theta, W1, b1, w2h, b2, out);
}
